// LegendreEncoding_7670811591192
// MI455X (gfx1250) — matrix operand order repaired and checked by reading, not compiled, not run on hardware
//
#include <hip/hip_runtime.h>
#include <hip/hip_bf16.h>

typedef __attribute__((ext_vector_type(16))) _Float16 v16h;
typedef __attribute__((ext_vector_type(8)))  float    v8f;
typedef __attribute__((ext_vector_type(8)))  int      v8i;

#define NTOT    131072   // N eigenvalues
#define KHEADS  16       // K heads
#define HALF    64       // H/2 active Legendre columns
#define WSTRIDE 129      // H+1 row stride of W

__device__ __forceinline__ float fast_tanhf(float x) {
#if __has_builtin(__builtin_amdgcn_tanhf)
    return __builtin_amdgcn_tanhf(x);   // native v_tanh_f32 on gfx1250
#else
    return tanhf(x);                    // fallback: ocml
#endif
}

// One wave32 per tile of 16 consecutive n-values; produces the full 16x16
// (n x k) output tile via 32 chained v_wmma_f32_16x16x32_f16 ops with a
// block-diagonal B. Contraction index j = k*64 + h (length 1024), chunked by
// 32 -> each WMMA step lies entirely inside one k-block, so B has a single
// nonzero column per step. Lane roles (wave32):
//   row  = lane & 15  -> A row M (n index) AND B/C/D column N (head k)
//   half = lane >> 4  -> K-chunk (A/B) and row-group (C/D) selector per ISA layout
__global__ void __launch_bounds__(256)
legendre_wmma_kernel(const float* __restrict__ e,
                     const float* __restrict__ W,
                     const float* __restrict__ b,
                     float* __restrict__ out) {
    const int lane = threadIdx.x & 31;
    const int wave = threadIdx.x >> 5;
    const int tile = blockIdx.x * 8 + wave;   // 8 waves per block
    const int n0   = tile * 16;

    const int row  = lane & 15;               // A-row (n) AND B/C/D column (head k)
    const int half = lane >> 4;               // 0 or 1
    const int kcol = row;                     // this lane's output head

    const float en = e[n0 + row];

    // const[k] = b[k] + W[k,0] + sum W[k,1..64]  (folded into C init)
    float ck = b[kcol] + W[kcol * WSTRIDE];
    {
        const float* w1 = W + kcol * WSTRIDE + 1;
        #pragma unroll
        for (int hh = 0; hh < HALF; ++hh) ck += w1[hh];
    }
    v8f c;
    #pragma unroll
    for (int r = 0; r < 8; ++r) c[r] = ck;    // all rows of column kcol share const[kcol]

    // --- Preload this lane's B values ONCE (no loads inside the k-loop). ---
    // B 32x16 f16, contiguous k order used here: element v -> row t = v + 16*half, column = lane&15.
    // Nonzero only in the two steps where k == kcol, with values
    // W2[kcol, hs*32 + t]. Two contiguous 16-float chunks per lane.
    const float* w2row = W + kcol * WSTRIDE + 1 + HALF;   // W2[kcol, 0..63]
    const int   toff   = half << 4;                       // 16*half
    v16h bm0, bm1;
    #pragma unroll
    for (int v = 0; v < 16; ++v) bm0[v] = (_Float16)w2row[toff + v];
    #pragma unroll
    for (int v = 0; v < 16; ++v) bm1[v] = (_Float16)w2row[32 + toff + v];
    // Keep packed-int views so the per-step mask lowers to v_cndmask_b32
    // (one op per f16 pair) instead of per-half v_cndmask_b16.
    const v8i bm0i = __builtin_bit_cast(v8i, bm0);
    const v8i bm1i = __builtin_bit_cast(v8i, bm1);

    // Per-lane A element -> contraction index t (matches B contiguous layout):
    //   t(v) = v + 16*half ; precompute (t+1)*0.2 arg scales.
    float f0[16];
    #pragma unroll
    for (int v = 0; v < 16; ++v) {
        const int t = v + (half << 4);
        f0[v] = (float)(t + 1) * 0.2f;
    }

    float p = 1.0f;
    #pragma unroll 1
    for (int k = 0; k < KHEADS; ++k) {
        p *= en;                               // en^(k+1)
        const bool active = (kcol == k);       // only column k of B nonzero

        #pragma unroll
        for (int hs = 0; hs < 2; ++hs) {
            const float hadd = hs ? 6.4f : 0.0f;   // +32 h-offset * 0.2

            // A: 16 tanh values for (row n0+row, head k, h = hs*32 + t(v))
            v16h a;
            #pragma unroll
            for (int v = 0; v < 16; ++v)
                a[v] = (_Float16)fast_tanhf(p * (f0[v] + hadd));

            // B: pure-register 32-bit select (8x v_cndmask_b32), no memory.
            v8i bsel;
            #pragma unroll
            for (int v = 0; v < 8; ++v)
                bsel[v] = active ? (hs ? bm1i[v] : bm0i[v]) : 0;
            const v16h bu = __builtin_bit_cast(v16h, bsel);

            c = __builtin_amdgcn_wmma_f32_16x16x32_f16(
                    false, a, false, bu, (short)0, c, false, false);
        }
    }

    // C/D layout: lane covers column kcol, rows M = r + 8*half.
    // out is [K, N] flat: out[k*NTOT + n]; 8 contiguous floats per lane.
    float* outk = out + (size_t)kcol * NTOT + n0 + (half << 3);
    #pragma unroll
    for (int r = 0; r < 8; ++r) outk[r] = c[r];
}

extern "C" void kernel_launch(void* const* d_in, const int* in_sizes, int n_in,
                              void* d_out, int out_size, void* d_ws, size_t ws_size,
                              hipStream_t stream) {
    (void)in_sizes; (void)n_in; (void)d_ws; (void)ws_size; (void)out_size;
    const float* e = (const float*)d_in[0];
    const float* W = (const float*)d_in[1];
    const float* b = (const float*)d_in[2];
    float* out = (float*)d_out;

    // 8 waves/block, one 16-row tile per wave: N/(16*8) = 1024 blocks.
    legendre_wmma_kernel<<<NTOT / (16 * 8), 256, 0, stream>>>(e, W, b, out);
}
